// FrameNetwork_9663676416053
// MI455X (gfx1250) — compile-verified, matrix operand order checked by reading, not run on hardware
//
#include <hip/hip_runtime.h>
#include <hip/hip_bf16.h>

#define Bsz 4
#define Nn 4096
#define Kn 24
#define BN (Bsz*Nn)
#define BNK (BN*Kn)
#define EPSf 1e-8f

typedef __attribute__((ext_vector_type(16))) __bf16 v16bf;
typedef __attribute__((ext_vector_type(8)))  float  v8f;

union Frag16 { v16bf v; unsigned short u[16]; uint4 q[2]; };

__device__ __forceinline__ unsigned short f2bf(float f) {
    union { float f; unsigned u; } x; x.f = f;
    unsigned r = x.u + 0x7fffu + ((x.u >> 16) & 1u);
    return (unsigned short)(r >> 16);
}
__device__ __forceinline__ float bf2f(unsigned short b) {
    union { unsigned u; float f; } x; x.u = ((unsigned)b) << 16; return x.f;
}
// K index inside a 16x32 16-bit WMMA fragment for (lane, element j)
__device__ __forceinline__ int k_of(int lane, int j) {
    int v = j >> 1, p = j & 1;
    int base = (lane >> 4) << 3;   // 0 or 8
    return (v < 4) ? (base + 2 * v + p) : (16 + base + 2 * (v - 4) + p);
}

// ---------------- zero init ----------------
__global__ void zero_kernel(float* a, size_t na, float* b, size_t nb) {
    size_t i = (size_t)blockIdx.x * blockDim.x + threadIdx.x;
    size_t stride = (size_t)gridDim.x * blockDim.x;
    for (size_t t = i; t < na; t += stride) a[t] = 0.f;
    for (size_t t = i; t < nb; t += stride) b[t] = 0.f;
}

// ---------------- weight pack: f32 [Kdim][Ndim] -> bf16 B-fragments ----------------
__global__ void pack_b_kernel(const float* __restrict__ W, int Kdim, int Ndim,
                              int KT, int NT, unsigned short* __restrict__ dst) {
    int t = blockIdx.x * blockDim.x + threadIdx.x;
    int total = KT * NT * 32;
    if (t >= total) return;
    int frag = t >> 5, lane = t & 31;
    int kt = frag / NT, nt = frag % NT;
    unsigned short* d = dst + (size_t)t * 16;
    for (int j = 0; j < 16; ++j) {
        int k = kt * 32 + k_of(lane, j);
        int n = nt * 16 + (lane & 15);
        float v = (k < Kdim && n < Ndim) ? W[(size_t)k * Ndim + n] : 0.f;
        d[j] = f2bf(v);
    }
}

// ---------------- kNN + edge geometry ----------------
__global__ __launch_bounds__(256) void knn_edge_kernel(
    const float* __restrict__ p, int* __restrict__ idxb,
    float* __restrict__ distb, float* __restrict__ evb) {
    __shared__ float P[Nn * 3];
    int b = blockIdx.x >> 4;          // 16 chunks per batch
    int chunk = blockIdx.x & 15;
    const float* pb = p + (size_t)b * Nn * 3;
    for (int t = threadIdx.x; t < Nn * 3; t += 256) P[t] = pb[t];
    __syncthreads();
    int i = chunk * 256 + threadIdx.x;
    float px = P[3 * i], py = P[3 * i + 1], pz = P[3 * i + 2];
    float bd[Kn]; int bi[Kn];
    for (int s = 0; s < Kn; ++s) { bd[s] = 3.0e38f; bi[s] = 0; }
    float worst = 3.0e38f; int wslot = 0;
    for (int j = 0; j < Nn; ++j) {
        if (j == i) continue;
        float dx = P[3 * j] - px, dy = P[3 * j + 1] - py, dz = P[3 * j + 2] - pz;
        float d2 = dx * dx + dy * dy + dz * dz;
        if (d2 < worst) {
            bd[wslot] = d2; bi[wslot] = j;
            worst = -1.f;
            for (int s = 0; s < Kn; ++s) if (bd[s] > worst) { worst = bd[s]; wslot = s; }
        }
    }
    size_t eb = ((size_t)b * Nn + i) * Kn;
    for (int kk = 0; kk < Kn; ++kk) {
        int j = bi[kk];
        idxb[eb + kk] = j;
        float dx = P[3 * j] - px, dy = P[3 * j + 1] - py, dz = P[3 * j + 2] - pz;
        float inv = rsqrtf(dx * dx + dy * dy + dz * dz + EPSf);
        evb[(eb + kk) * 3 + 0] = dx * inv;
        evb[(eb + kk) * 3 + 1] = dy * inv;
        evb[(eb + kk) * 3 + 2] = dz * inv;
        distb[eb + kk] = sqrtf(bd[kk] + EPSf);
    }
}

// ---------------- graph conv (one node per 128-thread block) ----------------
__global__ __launch_bounds__(128) void gconv_kernel(
    const float* __restrict__ s_in, const float* __restrict__ v_in,
    float* __restrict__ s_out, float* __restrict__ v_out,
    const int* __restrict__ idxb, const float* __restrict__ distb, const float* __restrict__ evb,
    const float* __restrict__ wh0, const float* __restrict__ wv0, const float* __restrict__ bs0,
    const float* __restrict__ wh1, const float* __restrict__ wv1, const float* __restrict__ bs1,
    const float* __restrict__ wh2, const float* __restrict__ wv2, const float* __restrict__ bs2,
    const unsigned short* __restrict__ wsp0, const unsigned short* __restrict__ wsp1,
    const unsigned short* __restrict__ wsp2,
    const float* __restrict__ lng, const float* __restrict__ lnb) {
    __shared__ alignas(16) unsigned short ABUF[32][384];  // per-edge scalar features (bf16)
    __shared__ unsigned short SBUF[32][128];              // per-edge s_out (bf16)
    __shared__ unsigned short VAb[24][33][3];             // per-edge vector features (bf16)
    __shared__ unsigned short VHb[24][33][3];             // vh temp (bf16)
    __shared__ float STMP[128];
    __shared__ float VTMP[16][3];
    __shared__ float RED[4];

    const int tid = threadIdx.x;
    const int lane = tid & 31, wave = tid >> 5;
    const int node = blockIdx.x;
    const int b = node / Nn;
    const size_t nbase = (size_t)node;
    const size_t ebase = nbase * Kn;

    // zero pad rows 24..31 once (WMMA rows are independent; pad outputs never read)
    for (int t = tid; t < 8 * 384; t += 128) ABUF[24 + t / 384][t % 384] = 0;

    // ---- stage 0: gather ms = [s_j | e_s | s_i], mv = [v_j | e_v | v_i]
    for (int t = tid; t < 24 * 128; t += 128) {
        int e = t >> 7, f = t & 127;
        int j = idxb[ebase + e];
        ABUF[e][f]       = f2bf(s_in[((size_t)b * Nn + j) * 128 + f]);
        ABUF[e][192 + f] = f2bf(s_in[nbase * 128 + f]);
    }
    {   // RBF edge scalars, computed on the fly from dist
        const float step = 20.0f / 61.0f;
        const float coeff = -0.5f / (step * step);
        for (int t = tid; t < 24 * 64; t += 128) {
            int e = t >> 6, f = t & 63;
            float d = distb[ebase + e] * 10.0f;
            float val;
            if (f == 0)       val = (d < 0.f) ? 1.f : 0.f;
            else if (f == 63) val = (d >= 20.f) ? 1.f : 0.f;
            else { float ov = 20.0f * (float)(f - 1) / 61.0f; float tt = d - ov; val = __expf(coeff * tt * tt); }
            ABUF[e][128 + f] = f2bf(val);
        }
    }
    // zero K-pad columns for layer 0 (cols 352..383; col 352 rewritten by vn in V1)
    for (int t = tid; t < 24 * 32; t += 128) {
        int e = t >> 5, f = 352 + (t & 31);
        ABUF[e][f] = 0;
    }
    for (int t = tid; t < 24 * 16 * 3; t += 128) {
        int e = t / 48, r = t % 48, i = r / 3, c = r % 3;
        int j = idxb[ebase + e];
        VAb[e][i][c]      = f2bf(v_in[((size_t)b * Nn + j) * 48 + i * 3 + c]);
        VAb[e][17 + i][c] = f2bf(v_in[nbase * 48 + i * 3 + c]);
    }
    for (int t = tid; t < 24 * 3; t += 128) {
        int e = t / 3, c = t % 3;
        VAb[e][16][c] = f2bf(evb[(ebase + e) * 3 + c]);
    }
    __syncthreads();

    const float* WHp[3] = { wh0, wh1, wh2 };
    const float* WVp[3] = { wv0, wv1, wv2 };
    const float* BSp[3] = { bs0, bs1, bs2 };
    const unsigned short* WSP[3] = { wsp0, wsp1, wsp2 };

    for (int L = 0; L < 3; ++L) {
        const int si   = (L == 0) ? 320 : 128;
        const int h    = (L == 0) ? 33  : 16;
        const int vi   = (L == 0) ? 33  : 16;
        const int ktN  = (L == 0) ? 12  : 5;
        const bool act = (L < 2);
        const float* wh = WHp[L]; const float* wv = WVp[L]; const float* bsv = BSp[L];
        const unsigned short* wsp = WSP[L];

        if (L > 0) {
            // s input = previous s_out; zero cols 144..159 (stale layer-0 data in K-pad range)
            for (int t = tid; t < 24 * 128; t += 128) { int e = t >> 7, f = t & 127; ABUF[e][f] = SBUF[e][f]; }
            for (int t = tid; t < 24 * 16; t += 128) { int e = t >> 4; ABUF[e][144 + (t & 15)] = 0; }
            __syncthreads();
        }

        // ---- V1: vh = wh @ v, vn -> ABUF[:, si..si+h)
        for (int t = tid; t < 24 * h; t += 128) {
            int e = t / h, hh = t % h;
            float a0 = 0, a1 = 0, a2 = 0;
            for (int i2 = 0; i2 < vi; ++i2) {
                float w = wh[hh * vi + i2];
                a0 += w * bf2f(VAb[e][i2][0]);
                a1 += w * bf2f(VAb[e][i2][1]);
                a2 += w * bf2f(VAb[e][i2][2]);
            }
            VHb[e][hh][0] = f2bf(a0); VHb[e][hh][1] = f2bf(a1); VHb[e][hh][2] = f2bf(a2);
            ABUF[e][si + hh] = f2bf(sqrtf(a0 * a0 + a1 * a1 + a2 * a2 + EPSf));
        }
        __syncthreads();

        // ---- scalar track: 2 M-tiles x 2 N-tiles per wave via WMMA, kt-outer loop
        {
            const int nt0 = wave * 2, nt1 = nt0 + 1;
            const int hlane = lane & 15;
            const int base = (lane >> 4) << 3;   // element offset 0 or 8
            float b0v = bsv[nt0 * 16 + hlane];
            float b1v = bsv[nt1 * 16 + hlane];
            v8f acc00, acc01, acc10, acc11;      // [tile][ntile]
#pragma unroll
            for (int r = 0; r < 8; ++r) { acc00[r] = b0v; acc01[r] = b1v; acc10[r] = b0v; acc11[r] = b1v; }
            const uint4* arow0 = (const uint4*)&ABUF[hlane][0];       // tile 0 row
            const uint4* arow1 = (const uint4*)&ABUF[16 + hlane][0];  // tile 1 row
            for (int kt = 0; kt < ktN; ++kt) {
                int kb = kt * 32;
                const uint4* bp0 = (const uint4*)(wsp + ((size_t)((kt * 8 + nt0) * 32 + lane)) * 16);
                const uint4* bp1 = (const uint4*)(wsp + ((size_t)((kt * 8 + nt1) * 32 + lane)) * 16);
                Frag16 fb0, fb1;
                fb0.q[0] = bp0[0]; fb0.q[1] = bp0[1];
                fb1.q[0] = bp1[0]; fb1.q[1] = bp1[1];
                int qi0 = (kb + base) >> 3;          // uint4 index of first 8-elem run
                int qi1 = (kb + base + 16) >> 3;     // second 8-elem run
                Frag16 fa0, fa1;
                fa0.q[0] = arow0[qi0]; fa0.q[1] = arow0[qi1];
                fa1.q[0] = arow1[qi0]; fa1.q[1] = arow1[qi1];
                acc00 = __builtin_amdgcn_wmma_f32_16x16x32_bf16(false, fa0.v, false, fb0.v, (short)0, acc00, false, false);
                acc01 = __builtin_amdgcn_wmma_f32_16x16x32_bf16(false, fa0.v, false, fb1.v, (short)0, acc01, false, false);
                acc10 = __builtin_amdgcn_wmma_f32_16x16x32_bf16(false, fa1.v, false, fb0.v, (short)0, acc10, false, false);
                acc11 = __builtin_amdgcn_wmma_f32_16x16x32_bf16(false, fa1.v, false, fb1.v, (short)0, acc11, false, false);
            }
#pragma unroll
            for (int r = 0; r < 8; ++r) {
                int rr0 = r + ((lane >> 4) << 3);
                int rr1 = 16 + rr0;
                float x00 = acc00[r], x01 = acc01[r], x10 = acc10[r], x11 = acc11[r];
                if (act) {
                    x00 = fmaxf(x00, 0.f); x01 = fmaxf(x01, 0.f);
                    x10 = fmaxf(x10, 0.f); x11 = fmaxf(x11, 0.f);
                }
                SBUF[rr0][nt0 * 16 + hlane] = f2bf(x00);
                SBUF[rr0][nt1 * 16 + hlane] = f2bf(x01);
                SBUF[rr1][nt0 * 16 + hlane] = f2bf(x10);
                SBUF[rr1][nt1 * 16 + hlane] = f2bf(x11);
            }
        }

        // ---- V2: v_out = wv @ vh (+ sigmoid gate), overwrite VA rows 0..15
        for (int t = tid; t < 24 * 16; t += 128) {
            int e = t >> 4, o = t & 15;
            float a0 = 0, a1 = 0, a2 = 0;
            for (int hh = 0; hh < h; ++hh) {
                float w = wv[o * h + hh];
                a0 += w * bf2f(VHb[e][hh][0]);
                a1 += w * bf2f(VHb[e][hh][1]);
                a2 += w * bf2f(VHb[e][hh][2]);
            }
            if (act) {
                float g = 1.f / (1.f + __expf(-sqrtf(a0 * a0 + a1 * a1 + a2 * a2 + EPSf)));
                a0 *= g; a1 *= g; a2 *= g;
            }
            VAb[e][o][0] = f2bf(a0); VAb[e][o][1] = f2bf(a1); VAb[e][o][2] = f2bf(a2);
        }
        __syncthreads();
    }

    // ---- mean over edges, residual, LayerNorm (s) + vector norm (v)
    {
        int f = tid;
        float acc = 0;
        for (int kk = 0; kk < 24; ++kk) acc += bf2f(SBUF[kk][f]);
        STMP[f] = s_in[nbase * 128 + f] + acc * (1.f / 24.f);
    }
    for (int t = tid; t < 48; t += 128) {
        int o = t / 3, c = t % 3;
        float acc = 0;
        for (int kk = 0; kk < 24; ++kk) acc += bf2f(VAb[kk][o][c]);
        VTMP[o][c] = v_in[nbase * 48 + o * 3 + c] + acc * (1.f / 24.f);
    }
    __syncthreads();
    if (tid == 0) {
        float s = 0, ss = 0;
        for (int f2 = 0; f2 < 128; ++f2) { float x = STMP[f2]; s += x; ss += x * x; }
        float mu = s / 128.f;
        RED[0] = mu; RED[1] = ss / 128.f - mu * mu;
    }
    if (tid == 1) {
        float s = 0;
        for (int o = 0; o < 16; ++o) for (int c = 0; c < 3; ++c) { float x = VTMP[o][c]; s += x * x; }
        RED[2] = s / 16.f;
    }
    __syncthreads();
    {
        int f = tid;
        float inv = rsqrtf(RED[1] + 1e-5f);
        s_out[nbase * 128 + f] = (STMP[f] - RED[0]) * inv * lng[f] + lnb[f];
    }
    {
        float invv = rsqrtf(RED[2] + EPSf);
        for (int t = tid; t < 48; t += 128) {
            int o = t / 3, c = t % 3;
            v_out[nbase * 48 + o * 3 + c] = VTMP[o][c] * invv;
        }
    }
}

// ---------------- output GVP + frames (16 nodes per wave) ----------------
__global__ __launch_bounds__(32) void out_kernel(
    const float* __restrict__ s_fin, const float* __restrict__ v_fin,
    const float* __restrict__ who, const float* __restrict__ wvo, const float* __restrict__ bso,
    const unsigned short* __restrict__ owsp,
    float* __restrict__ Rout, float* __restrict__ ysout) {
    __shared__ alignas(16) unsigned short AB2[16][160];
    __shared__ float VH2[16][16][3];
    int lane = threadIdx.x;
    size_t nb = (size_t)blockIdx.x * 16;
    for (int t = lane; t < 16 * 128; t += 32) {
        int m = t >> 7, f = t & 127;
        AB2[m][f] = f2bf(s_fin[(nb + m) * 128 + f]);
    }
    for (int t = lane; t < 16 * 16; t += 32) {   // zero K-pad cols 144..159
        int m = t >> 4;
        AB2[m][144 + (t & 15)] = 0;
    }
    for (int t = lane; t < 16 * 16; t += 32) {
        int m = t >> 4, hh = t & 15;
        float a0 = 0, a1 = 0, a2 = 0;
        for (int i = 0; i < 16; ++i) {
            float w = who[hh * 16 + i];
            const float* vp = v_fin + (nb + m) * 48 + i * 3;
            a0 += w * vp[0]; a1 += w * vp[1]; a2 += w * vp[2];
        }
        VH2[m][hh][0] = a0; VH2[m][hh][1] = a1; VH2[m][hh][2] = a2;
        AB2[m][128 + hh] = f2bf(sqrtf(a0 * a0 + a1 * a1 + a2 * a2 + EPSf));
    }
    __syncthreads();
    {
        const int hlane = lane & 15;
        const int base = (lane >> 4) << 3;
        const uint4* arow = (const uint4*)&AB2[hlane][0];
        for (int nt = 0; nt < 8; ++nt) {
            float bv = bso[nt * 16 + hlane];
            v8f acc;
#pragma unroll
            for (int r = 0; r < 8; ++r) acc[r] = bv;
            for (int kt = 0; kt < 5; ++kt) {
                int kb = kt * 32;
                Frag16 fa, fb;
                fa.q[0] = arow[(kb + base) >> 3];
                fa.q[1] = arow[(kb + base + 16) >> 3];
                const uint4* bp = (const uint4*)(owsp + ((size_t)((kt * 8 + nt) * 32 + lane)) * 16);
                fb.q[0] = bp[0]; fb.q[1] = bp[1];
                acc = __builtin_amdgcn_wmma_f32_16x16x32_bf16(false, fa.v, false, fb.v, (short)0, acc, false, false);
            }
#pragma unroll
            for (int r = 0; r < 8; ++r) {
                int m = r + ((lane >> 4) << 3);
                ysout[(nb + m) * 128 + nt * 16 + hlane] = acc[r];
            }
        }
    }
    if (lane < 16) {
        int m = lane;
        float a[3] = {0, 0, 0}, c[3] = {0, 0, 0};
        for (int hh = 0; hh < 16; ++hh) {
            float w0 = wvo[hh], w1 = wvo[16 + hh];
            a[0] += w0 * VH2[m][hh][0]; a[1] += w0 * VH2[m][hh][1]; a[2] += w0 * VH2[m][hh][2];
            c[0] += w1 * VH2[m][hh][0]; c[1] += w1 * VH2[m][hh][1]; c[2] += w1 * VH2[m][hh][2];
        }
        float inva = rsqrtf(a[0] * a[0] + a[1] * a[1] + a[2] * a[2] + EPSf);
        float e1[3] = { a[0] * inva, a[1] * inva, a[2] * inva };
        float dd = c[0] * e1[0] + c[1] * e1[1] + c[2] * e1[2];
        float u2[3] = { c[0] - dd * e1[0], c[1] - dd * e1[1], c[2] - dd * e1[2] };
        float invu = rsqrtf(u2[0] * u2[0] + u2[1] * u2[1] + u2[2] * u2[2] + EPSf);
        float e2[3] = { u2[0] * invu, u2[1] * invu, u2[2] * invu };
        float e3[3] = { e1[1] * e2[2] - e1[2] * e2[1],
                        e1[2] * e2[0] - e1[0] * e2[2],
                        e1[0] * e2[1] - e1[1] * e2[0] };
        float* Rp = Rout + (nb + m) * 9;
        Rp[0] = e1[0]; Rp[1] = e2[0]; Rp[2] = e3[0];
        Rp[3] = e1[1]; Rp[4] = e2[1]; Rp[5] = e3[1];
        Rp[6] = e1[2]; Rp[7] = e2[2]; Rp[8] = e3[2];
    }
}

// ---------------- host ----------------
extern "C" void kernel_launch(void* const* d_in, const int* in_sizes, int n_in,
                              void* d_out, int out_size, void* d_ws, size_t ws_size,
                              hipStream_t stream) {
    (void)in_sizes; (void)n_in; (void)out_size; (void)ws_size;
    const float* p = (const float*)d_in[0];

    char* wsb = (char*)d_ws;
    size_t off = 0;
    auto alloc = [&](size_t bytes) -> char* {
        char* r = wsb + off; off = (off + bytes + 255) & ~(size_t)255; return r;
    };
    int* idxb   = (int*)alloc((size_t)BNK * 4);
    float* distb = (float*)alloc((size_t)BNK * 4);
    float* evb  = (float*)alloc((size_t)BNK * 12);
    float* bufS[2], * bufV[2];
    bufS[0] = (float*)alloc((size_t)BN * 128 * 4);
    bufS[1] = (float*)alloc((size_t)BN * 128 * 4);
    bufV[0] = (float*)alloc((size_t)BN * 48 * 4);
    bufV[1] = (float*)alloc((size_t)BN * 48 * 4);
    unsigned short* wsp[3][3];
    for (int g = 0; g < 3; ++g) {
        wsp[g][0] = (unsigned short*)alloc((size_t)12 * 8 * 32 * 16 * 2);
        wsp[g][1] = (unsigned short*)alloc((size_t)5 * 8 * 32 * 16 * 2);
        wsp[g][2] = (unsigned short*)alloc((size_t)5 * 8 * 32 * 16 * 2);
    }
    unsigned short* owsp = (unsigned short*)alloc((size_t)5 * 8 * 32 * 16 * 2);

    zero_kernel<<<512, 256, 0, stream>>>(bufS[0], (size_t)BN * 128, bufV[0], (size_t)BN * 48);

    for (int g = 0; g < 3; ++g) {
        int b0 = 1 + g * 14;   // ln_b, ln_g, then msg0..2 as (bs, wh, ws, wv)
        pack_b_kernel<<<(12 * 8 * 32 + 127) / 128, 128, 0, stream>>>((const float*)d_in[b0 + 4], 353, 128, 12, 8, wsp[g][0]);
        pack_b_kernel<<<(5 * 8 * 32 + 127) / 128, 128, 0, stream>>>((const float*)d_in[b0 + 8], 144, 128, 5, 8, wsp[g][1]);
        pack_b_kernel<<<(5 * 8 * 32 + 127) / 128, 128, 0, stream>>>((const float*)d_in[b0 + 12], 144, 128, 5, 8, wsp[g][2]);
    }
    pack_b_kernel<<<(5 * 8 * 32 + 127) / 128, 128, 0, stream>>>((const float*)d_in[45], 144, 128, 5, 8, owsp);

    knn_edge_kernel<<<Bsz * (Nn / 256), 256, 0, stream>>>(p, idxb, distb, evb);

    for (int g = 0; g < 3; ++g) {
        int b0 = 1 + g * 14;
        gconv_kernel<<<BN, 128, 0, stream>>>(
            bufS[g & 1], bufV[g & 1], bufS[(g + 1) & 1], bufV[(g + 1) & 1],
            idxb, distb, evb,
            (const float*)d_in[b0 + 3], (const float*)d_in[b0 + 5], (const float*)d_in[b0 + 2],
            (const float*)d_in[b0 + 7], (const float*)d_in[b0 + 9], (const float*)d_in[b0 + 6],
            (const float*)d_in[b0 + 11], (const float*)d_in[b0 + 13], (const float*)d_in[b0 + 10],
            wsp[g][0], wsp[g][1], wsp[g][2],
            (const float*)d_in[b0 + 1], (const float*)d_in[b0 + 0]);
    }

    float* Rout = (float*)d_out;
    float* ysout = Rout + (size_t)BN * 9;
    out_kernel<<<BN / 16, 32, 0, stream>>>(
        bufS[1], bufV[1],
        (const float*)d_in[44], (const float*)d_in[46], (const float*)d_in[43],
        owsp, Rout, ysout);
}
